// CausalSelfAttention_17617955848355
// MI455X (gfx1250) — matrix operand order repaired and checked by reading, not compiled, not run on hardware
//
#include <hip/hip_runtime.h>
#include <hip/hip_bf16.h>

// Problem constants (from the reference)
#define BSZ 2
#define TT  2048
#define CC  1024
#define HH  16
#define DD  64
#define N3  3072     // 3*C
#define MM  (BSZ*TT) // 4096

typedef __attribute__((ext_vector_type(16))) __bf16 v16bf;
typedef __attribute__((ext_vector_type(8)))  __bf16 v8bf;
typedef __attribute__((ext_vector_type(8)))  float  v8f;

union AFrag { v16bf v; v8bf h[2]; __bf16 e[16]; };
union CFrag { v8f  v; float  e[8];  };
union BF8   { v8bf v; __bf16 e[8]; };
union F4    { float4 v; float f[4]; };

__device__ __forceinline__ __bf16 f2bf(float f) {
    return static_cast<__bf16>(f);   // hardware cvt if available on gfx1250
}
__device__ __forceinline__ unsigned pack_bf2(__bf16 a, __bf16 b) {
    return (unsigned)__builtin_bit_cast(unsigned short, a) |
           ((unsigned)__builtin_bit_cast(unsigned short, b) << 16);
}

// ---------------------------------------------------------------------------
// Kernel 0: one-time fp32 -> bf16 conversion (bandwidth-bound, ~negligible).
// ---------------------------------------------------------------------------
__global__ __launch_bounds__(256) void cvt_kernel(
    const float* __restrict__ src, __bf16* __restrict__ dst, int n8)
{
    int i = blockIdx.x * blockDim.x + threadIdx.x;   // one v8bf per thread
    if (i >= n8) return;
    const float4* s = (const float4*)(src + (size_t)i * 8);
    F4 f0, f1; f0.v = s[0]; f1.v = s[1];
    BF8 o;
    #pragma unroll
    for (int j = 0; j < 4; ++j) { o.e[j] = f2bf(f0.f[j]); o.e[4+j] = f2bf(f1.f[j]); }
    *(v8bf*)(dst + (size_t)i * 8) = o.v;
}

// ---------------------------------------------------------------------------
// Kernel 1: qkv = xb @ Wab + b_attn -> head-major bf16 Q/K/V. All-bf16 LDS
// staging (b128 copies), double-buffered, batched fragments.
// ---------------------------------------------------------------------------
__global__ __launch_bounds__(128) void qkv_kernel(
    const __bf16* __restrict__ xb, const __bf16* __restrict__ Wab,
    const float* __restrict__ ba,
    __bf16* __restrict__ Qb, __bf16* __restrict__ Kb, __bf16* __restrict__ Vb)
{
    __shared__ __attribute__((aligned(16))) __bf16 sA [2][64][40]; // [m][k]
    __shared__ __attribute__((aligned(16))) __bf16 sBT[2][64][40]; // [n][k]

    const int tid  = threadIdx.x;
    const int lane = tid & 31;
    const int w    = tid >> 5;
    const int hi   = lane >> 4;
    const int lo   = lane & 15;
    const int mBase = blockIdx.x * 64;
    const int nBase = blockIdx.y * 64;

    const int arow = tid >> 1, acb = (tid & 1) * 16;      // A staging map
    const int bk0  = (tid >> 3) * 2, bnb = (tid & 7) * 8; // B staging map

    CFrag acc[4];
    #pragma unroll
    for (int i = 0; i < 4; ++i)
        #pragma unroll
        for (int r = 0; r < 8; ++r) acc[i].e[r] = 0.0f;

    { // prologue: stage chunk 0 into buffer 0
        const v8bf* as = (const v8bf*)(xb + (size_t)(mBase + arow) * CC + acb);
        *(v8bf*)&sA[0][arow][acb]     = as[0];
        *(v8bf*)&sA[0][arow][acb + 8] = as[1];
        BF8 w0, w1;
        w0.v = *(const v8bf*)(Wab + (size_t)bk0 * N3 + nBase + bnb);
        w1.v = *(const v8bf*)(Wab + (size_t)(bk0 + 1) * N3 + nBase + bnb);
        #pragma unroll
        for (int j = 0; j < 8; ++j)
            *(unsigned*)&sBT[0][bnb + j][bk0] = pack_bf2(w0.e[j], w1.e[j]);
    }
    __syncthreads();

    int cur = 0;
    for (int kk = 0; kk < CC; kk += 32) {
        const bool more = (kk + 32) < CC;
        v8bf anx0, anx1; BF8 wnx0, wnx1;
        if (more) {  // register-stage next chunk (overlaps with WMMAs below)
            const v8bf* as = (const v8bf*)(xb + (size_t)(mBase + arow) * CC + kk + 32 + acb);
            anx0 = as[0]; anx1 = as[1];
            wnx0.v = *(const v8bf*)(Wab + (size_t)(kk + 32 + bk0) * N3 + nBase + bnb);
            wnx1.v = *(const v8bf*)(Wab + (size_t)(kk + 32 + bk0 + 1) * N3 + nBase + bnb);
        }

        AFrag a, bf[4];   // batched fragment loads, then 4 WMMAs
        a.h[0] = *(const v8bf*)&sA[cur][w * 16 + lo][hi * 16];
        a.h[1] = *(const v8bf*)&sA[cur][w * 16 + lo][hi * 16 + 8];
        #pragma unroll
        for (int nt = 0; nt < 4; ++nt)
            bf[nt].v = *(const v16bf*)&sBT[cur][nt * 16 + lo][hi << 4];
        #pragma unroll
        for (int nt = 0; nt < 4; ++nt)
            acc[nt].v = __builtin_amdgcn_wmma_f32_16x16x32_bf16(
                false, a.v, false, bf[nt].v, (short)0, acc[nt].v, false, false);

        if (more) {  // store next chunk into the other buffer
            *(v8bf*)&sA[cur ^ 1][arow][acb]     = anx0;
            *(v8bf*)&sA[cur ^ 1][arow][acb + 8] = anx1;
            #pragma unroll
            for (int j = 0; j < 8; ++j)
                *(unsigned*)&sBT[cur ^ 1][bnb + j][bk0] = pack_bf2(wnx0.e[j], wnx1.e[j]);
        }
        __syncthreads();
        cur ^= 1;
    }

    // epilogue: sel and h are uniform per block (nBase is 64-aligned)
    const int sel = nBase >> 10;                 // 0=q 1=k 2=v
    const int h   = (nBase & 1023) >> 6;
    __bf16* dst = (sel == 0) ? Qb : ((sel == 1) ? Kb : Vb);
    #pragma unroll
    for (int nt = 0; nt < 4; ++nt) {
        #pragma unroll
        for (int r = 0; r < 8; ++r) {
            int m = mBase + w * 16 + r + (hi << 3);
            int n = nBase + nt * 16 + lo;
            float val = acc[nt].e[r] + ba[n];
            int b = m >> 11, t = m & 2047;
            size_t idx = ((size_t)((b * HH + h) * TT + t)) * DD + nt * 16 + lo;
            dst[idx] = f2bf(val);
        }
    }
}

// ---------------------------------------------------------------------------
// Kernel 2: causal flash attention, double-buffered K/V tiles + prefetch.
// ---------------------------------------------------------------------------
__global__ __launch_bounds__(128) void attn_kernel(
    const __bf16* __restrict__ Qb, const __bf16* __restrict__ Kb,
    const __bf16* __restrict__ Vb, __bf16* __restrict__ Yb)
{
    __shared__ __attribute__((aligned(16))) __bf16 sK [2][32][72]; // [keypos][feat]
    __shared__ __attribute__((aligned(16))) __bf16 sVT[2][64][40]; // [feat][keypos]
    __shared__ __attribute__((aligned(16))) __bf16 sP [4][16][40];

    const int tid  = threadIdx.x;
    const int lane = tid & 31;
    const int w    = tid >> 5;
    const int hi   = lane >> 4;
    const int lo   = lane & 15;
    const int bh   = blockIdx.x;
    const int qblk = blockIdx.y;
    const int q0   = qblk * 64 + w * 16;
    const float scale = 0.125f;               // 1/sqrt(64)

    const __bf16* Qh = Qb + (size_t)bh * TT * DD;
    const __bf16* Kh = Kb + (size_t)bh * TT * DD;
    const __bf16* Vh = Vb + (size_t)bh * TT * DD;

    const int krow = tid >> 2, kfb = (tid & 3) * 16;      // K staging map
    const int vk0  = (tid >> 3) * 2, vfb = (tid & 7) * 8; // V staging map

    // Q fragments: two K=32 chunks over D=64, 128-bit global loads
    AFrag qa[2];
    #pragma unroll
    for (int kc = 0; kc < 2; ++kc) {
        const __bf16* qrow = Qh + (size_t)(q0 + lo) * DD + kc * 32;
        qa[kc].h[0] = *(const v8bf*)(qrow + hi * 16);
        qa[kc].h[1] = *(const v8bf*)(qrow + hi * 16 + 8);
    }

    CFrag o[4];
    #pragma unroll
    for (int i = 0; i < 4; ++i)
        #pragma unroll
        for (int r = 0; r < 8; ++r) o[i].e[r] = 0.0f;
    float mrow[8], lrow[8];
    #pragma unroll
    for (int r = 0; r < 8; ++r) { mrow[r] = -3.0e38f; lrow[r] = 0.0f; }

    { // prologue: stage tile 0 into buffer 0
        const v8bf* ks = (const v8bf*)(Kh + (size_t)krow * DD + kfb);
        *(v8bf*)&sK[0][krow][kfb]     = ks[0];
        *(v8bf*)&sK[0][krow][kfb + 8] = ks[1];
        BF8 v0, v1;
        v0.v = *(const v8bf*)(Vh + (size_t)vk0 * DD + vfb);
        v1.v = *(const v8bf*)(Vh + (size_t)(vk0 + 1) * DD + vfb);
        #pragma unroll
        for (int j = 0; j < 8; ++j)
            *(unsigned*)&sVT[0][vfb + j][vk0] = pack_bf2(v0.e[j], v1.e[j]);
    }
    __syncthreads();

    int cur = 0;
    const int ntiles = qblk * 2 + 2;
    for (int kt = 0; kt < ntiles; ++kt) {
        const int kbase = kt * 32;
        const bool more = (kt + 1) < ntiles;

        if (kt + 2 < ntiles) {   // prefetch tile kt+2 (global_prefetch_b8)
            __builtin_prefetch(Kh + (size_t)(kbase + 64 + krow) * DD, 0, 1);
            __builtin_prefetch(Vh + (size_t)(kbase + 64 + krow) * DD, 0, 1);
        }

        v8bf knx0, knx1; BF8 vnx0, vnx1;
        if (more) {  // register-stage next K/V tile
            const v8bf* ks = (const v8bf*)(Kh + (size_t)(kbase + 32 + krow) * DD + kfb);
            knx0 = ks[0]; knx1 = ks[1];
            vnx0.v = *(const v8bf*)(Vh + (size_t)(kbase + 32 + vk0) * DD + vfb);
            vnx1.v = *(const v8bf*)(Vh + (size_t)(kbase + 32 + vk0 + 1) * DD + vfb);
        }

        // scores S = Q K^T : batch 4 B-fragments then 4 WMMAs
        AFrag bq[4];
        #pragma unroll
        for (int nt = 0; nt < 2; ++nt)
            #pragma unroll
            for (int kc = 0; kc < 2; ++kc)
                bq[nt * 2 + kc].v =
                    *(const v16bf*)&sK[cur][nt * 16 + lo][kc * 32 + (hi << 4)];
        CFrag s[2];
        #pragma unroll
        for (int nt = 0; nt < 2; ++nt) {
            #pragma unroll
            for (int r = 0; r < 8; ++r) s[nt].e[r] = 0.0f;
            #pragma unroll
            for (int kc = 0; kc < 2; ++kc)
                s[nt].v = __builtin_amdgcn_wmma_f32_16x16x32_bf16(
                    false, qa[kc].v, false, bq[nt * 2 + kc].v, (short)0,
                    s[nt].v, false, false);
            int kg = kbase + nt * 16 + lo;
            #pragma unroll
            for (int r = 0; r < 8; ++r) {      // scale + causal mask (select)
                int qg = q0 + r + (hi << 3);
                s[nt].e[r] = (kg <= qg) ? s[nt].e[r] * scale : -3.0e38f;
            }
        }

        // online softmax (row reductions across each 16-lane half) + P store
        #pragma unroll
        for (int r = 0; r < 8; ++r) {
            float v = fmaxf(s[0].e[r], s[1].e[r]);
            #pragma unroll
            for (int off = 1; off < 16; off <<= 1)
                v = fmaxf(v, __shfl_xor(v, off, 32));
            float mnew  = fmaxf(mrow[r], v);
            float alpha = __expf(mrow[r] - mnew);
            float p0 = __expf(s[0].e[r] - mnew);
            float p1 = __expf(s[1].e[r] - mnew);
            float psum = p0 + p1;
            #pragma unroll
            for (int off = 1; off < 16; off <<= 1)
                psum += __shfl_xor(psum, off, 32);
            lrow[r] = lrow[r] * alpha + psum;
            mrow[r] = mnew;
            #pragma unroll
            for (int nt4 = 0; nt4 < 4; ++nt4) o[nt4].e[r] *= alpha;
            int row = r + (hi << 3);
            sP[w][row][lo]      = f2bf(p0);
            sP[w][row][16 + lo] = f2bf(p1);
        }
        asm volatile("s_wait_dscnt 0" ::: "memory"); // wave-local LDS ordering

        // O += P V : batched fragment loads then 4 WMMAs
        AFrag pa, bv[4];
        pa.h[0] = *(const v8bf*)&sP[w][lo][hi * 16];
        pa.h[1] = *(const v8bf*)&sP[w][lo][hi * 16 + 8];
        #pragma unroll
        for (int nt4 = 0; nt4 < 4; ++nt4)
            bv[nt4].v = *(const v16bf*)&sVT[cur][nt4 * 16 + lo][hi << 4];
        #pragma unroll
        for (int nt4 = 0; nt4 < 4; ++nt4)
            o[nt4].v = __builtin_amdgcn_wmma_f32_16x16x32_bf16(
                false, pa.v, false, bv[nt4].v, (short)0, o[nt4].v, false, false);

        if (more) {  // store next tile into the other buffer
            *(v8bf*)&sK[cur ^ 1][krow][kfb]     = knx0;
            *(v8bf*)&sK[cur ^ 1][krow][kfb + 8] = knx1;
            #pragma unroll
            for (int j = 0; j < 8; ++j)
                *(unsigned*)&sVT[cur ^ 1][vfb + j][vk0] = pack_bf2(vnx0.e[j], vnx1.e[j]);
        }
        __syncthreads();
        cur ^= 1;
    }

    // epilogue: Y[b, t, h*64 + d] = O / l
    int b = bh / HH, h = bh % HH;
    #pragma unroll
    for (int nt4 = 0; nt4 < 4; ++nt4) {
        #pragma unroll
        for (int r = 0; r < 8; ++r) {
            int t = q0 + r + (hi << 3);
            int d = nt4 * 16 + lo;
            float val = o[nt4].e[r] / lrow[r];
            Yb[((size_t)(b * TT + t)) * CC + h * DD + d] = f2bf(val);
        }
    }
}

// ---------------------------------------------------------------------------
// Kernel 3: out = Y @ Wpb + b_proj  (all-bf16 staging; fp32 out)
// ---------------------------------------------------------------------------
__global__ __launch_bounds__(128) void proj_kernel(
    const __bf16* __restrict__ Yb, const __bf16* __restrict__ Wpb,
    const float* __restrict__ bp, float* __restrict__ out)
{
    __shared__ __attribute__((aligned(16))) __bf16 sA [2][64][40];
    __shared__ __attribute__((aligned(16))) __bf16 sBT[2][64][40];

    const int tid  = threadIdx.x;
    const int lane = tid & 31;
    const int w    = tid >> 5;
    const int hi   = lane >> 4;
    const int lo   = lane & 15;
    const int mBase = blockIdx.x * 64;
    const int nBase = blockIdx.y * 64;

    const int arow = tid >> 1, acb = (tid & 1) * 16;
    const int bk0  = (tid >> 3) * 2, bnb = (tid & 7) * 8;

    CFrag acc[4];
    #pragma unroll
    for (int i = 0; i < 4; ++i)
        #pragma unroll
        for (int r = 0; r < 8; ++r) acc[i].e[r] = 0.0f;

    { // prologue: stage chunk 0
        const v8bf* as = (const v8bf*)(Yb + (size_t)(mBase + arow) * CC + acb);
        *(v8bf*)&sA[0][arow][acb]     = as[0];
        *(v8bf*)&sA[0][arow][acb + 8] = as[1];
        BF8 w0, w1;
        w0.v = *(const v8bf*)(Wpb + (size_t)bk0 * CC + nBase + bnb);
        w1.v = *(const v8bf*)(Wpb + (size_t)(bk0 + 1) * CC + nBase + bnb);
        #pragma unroll
        for (int j = 0; j < 8; ++j)
            *(unsigned*)&sBT[0][bnb + j][bk0] = pack_bf2(w0.e[j], w1.e[j]);
    }
    __syncthreads();

    int cur = 0;
    for (int kk = 0; kk < CC; kk += 32) {
        const bool more = (kk + 32) < CC;
        v8bf anx0, anx1; BF8 wnx0, wnx1;
        if (more) {
            const v8bf* as = (const v8bf*)(Yb + (size_t)(mBase + arow) * CC + kk + 32 + acb);
            anx0 = as[0]; anx1 = as[1];
            wnx0.v = *(const v8bf*)(Wpb + (size_t)(kk + 32 + bk0) * CC + nBase + bnb);
            wnx1.v = *(const v8bf*)(Wpb + (size_t)(kk + 32 + bk0 + 1) * CC + nBase + bnb);
        }

        AFrag a, bf[4];
        a.h[0] = *(const v8bf*)&sA[cur][w * 16 + lo][hi * 16];
        a.h[1] = *(const v8bf*)&sA[cur][w * 16 + lo][hi * 16 + 8];
        #pragma unroll
        for (int nt = 0; nt < 4; ++nt)
            bf[nt].v = *(const v16bf*)&sBT[cur][nt * 16 + lo][hi << 4];
        #pragma unroll
        for (int nt = 0; nt < 4; ++nt)
            acc[nt].v = __builtin_amdgcn_wmma_f32_16x16x32_bf16(
                false, a.v, false, bf[nt].v, (short)0, acc[nt].v, false, false);

        if (more) {
            *(v8bf*)&sA[cur ^ 1][arow][acb]     = anx0;
            *(v8bf*)&sA[cur ^ 1][arow][acb + 8] = anx1;
            #pragma unroll
            for (int j = 0; j < 8; ++j)
                *(unsigned*)&sBT[cur ^ 1][bnb + j][bk0] = pack_bf2(wnx0.e[j], wnx1.e[j]);
        }
        __syncthreads();
        cur ^= 1;
    }

    #pragma unroll
    for (int nt = 0; nt < 4; ++nt) {
        #pragma unroll
        for (int r = 0; r < 8; ++r) {
            int m = mBase + w * 16 + r + (hi << 3);
            int n = nBase + nt * 16 + lo;
            out[(size_t)m * CC + n] = acc[nt].e[r] + bp[n];
        }
    }
}

// ---------------------------------------------------------------------------
extern "C" void kernel_launch(void* const* d_in, const int* in_sizes, int n_in,
                              void* d_out, int out_size, void* d_ws, size_t ws_size,
                              hipStream_t stream) {
    const float* x  = (const float*)d_in[0];
    const float* Wa = (const float*)d_in[1];
    const float* ba = (const float*)d_in[2];
    const float* Wp = (const float*)d_in[3];
    const float* bp = (const float*)d_in[4];
    float* out = (float*)d_out;

    const size_t S   = (size_t)BSZ * HH * TT * DD;  // 4,194,304
    const size_t NX  = (size_t)MM * CC;             // 4,194,304 (x)
    const size_t NWA = (size_t)CC * N3;             // 3,145,728
    const size_t NWP = (size_t)CC * CC;             // 1,048,576

    __bf16* Qb  = (__bf16*)d_ws;
    __bf16* Kb  = Qb + S;
    __bf16* Vb  = Kb + S;
    __bf16* Yb  = Vb + S;
    __bf16* xbf = Yb + S;
    __bf16* Wab = xbf + NX;
    __bf16* Wpb = Wab + NWA;                        // total ~48.2 MB

    cvt_kernel<<<(int)(NX  / 8 + 255) / 256, 256, 0, stream>>>(x,  xbf, (int)(NX  / 8));
    cvt_kernel<<<(int)(NWA / 8 + 255) / 256, 256, 0, stream>>>(Wa, Wab, (int)(NWA / 8));
    cvt_kernel<<<(int)(NWP / 8 + 255) / 256, 256, 0, stream>>>(Wp, Wpb, (int)(NWP / 8));

    qkv_kernel <<<dim3(MM / 64, N3 / 64), 128, 0, stream>>>(xbf, Wab, ba, Qb, Kb, Vb);
    attn_kernel<<<dim3(BSZ * HH, TT / 64), 128, 0, stream>>>(Qb, Kb, Vb, Yb);
    proj_kernel<<<dim3(MM / 64, CC / 64), 128, 0, stream>>>(Yb, Wpb, bp, out);
}
